// RNNLM_Model_28372553957785
// MI455X (gfx1250) — compile-verified, matrix operand order checked by reading, not run on hardware
//
#include <hip/hip_runtime.h>

#define B_   32
#define T_   128
#define V_   32000
#define E_   256
#define HID_ 512

typedef __attribute__((ext_vector_type(2)))  float    v2f;
typedef __attribute__((ext_vector_type(8)))  float    v8f;
typedef __attribute__((ext_vector_type(4)))  int      v4i;
typedef __attribute__((ext_vector_type(8)))  _Float16 v8h;
typedef __attribute__((ext_vector_type(16))) _Float16 v16h;

#if defined(__HIP_DEVICE_COMPILE__) && __has_builtin(__builtin_amdgcn_global_load_async_to_lds_b128)
#define HAVE_ASYNC_LDS 1
#else
#define HAVE_ASYNC_LDS 0
#endif

#if HAVE_ASYNC_LDS
typedef __attribute__((address_space(1))) v4i* gv4i_p;   // global b128 pointer
typedef __attribute__((address_space(3))) v4i* lv4i_p;   // LDS b128 pointer

__device__ __forceinline__ void async_copy_b128(const _Float16* g, _Float16* l) {
    __builtin_amdgcn_global_load_async_to_lds_b128(
        (gv4i_p)g, (lv4i_p)l, 0, 0);
}
#if __has_builtin(__builtin_amdgcn_s_wait_asynccnt)
#define WAIT_ASYNC(n) __builtin_amdgcn_s_wait_asynccnt(n)
#else
#define WAIT_ASYNC(n) asm volatile("s_wait_asynccnt %0" :: "i"(n) : "memory")
#endif
#endif

// ---------------------------------------------------------------------------
// Kernel 1: convert U [512,32000] f32 -> f16 in WMMA-B-fragment-major layout.
// Uh[ntile*8192 + kstep*512 + lane*16 + e]; element e of lane L covers
// K = (e%8) + (L/16)*8 + (e/8)*16 within a 32-K step, n = ntile*16 + L%16.
// One workgroup per 16-column ntile, staged through LDS (coalesced R+W).
// ---------------------------------------------------------------------------
__global__ __launch_bounds__(256) void k_convert_U(const float* __restrict__ U,
                                                   _Float16* __restrict__ Uh) {
    __shared__ _Float16 lds[512 * 16];           // [k][nsub], 16 KB
    const int ntile = blockIdx.x;                // 0..1999
    const int tid   = threadIdx.x;

    #pragma unroll
    for (int i = 0; i < 32; ++i) {
        int pos  = i * 256 + tid;                // 0..8191
        int k    = pos >> 4;
        int nsub = pos & 15;
        lds[pos] = (_Float16)U[k * V_ + ntile * 16 + nsub];
    }
    __syncthreads();

    alignas(32) _Float16 tmp[32];
    const int base = tid * 32;                   // 64-byte aligned output chunk
    #pragma unroll
    for (int i = 0; i < 32; ++i) {
        int pos   = base + i;
        int kstep = pos >> 9;
        int rem   = pos & 511;
        int lane  = rem >> 4;
        int e     = rem & 15;
        int k     = kstep * 32 + (e & 7) + ((lane >> 4) << 3) + ((e >> 3) << 4);
        tmp[i]    = lds[k * 16 + (lane & 15)];
    }
    v16h* dst = (v16h*)(Uh + (size_t)ntile * 8192 + base);
    dst[0] = *(const v16h*)&tmp[0];
    dst[1] = *(const v16h*)&tmp[16];
}

// ---------------------------------------------------------------------------
// Kernel 1b: swizzle H [512,512] f32 into f32-WMMA B-fragment-major layout so
// each lane's per-step fragment is one contiguous b64:
// Hf[((ntile*128 + kk)*32 + lane)*2 + d] = H[(kk*4 + (lane/16)*2 + d)*512 + ntile*16 + lane%16]
// ---------------------------------------------------------------------------
__global__ __launch_bounds__(256) void k_convert_H(const float* __restrict__ H,
                                                   float* __restrict__ Hf) {
    const int idx   = blockIdx.x * 256 + threadIdx.x; // 0..131071
    const int lane  = idx & 31;
    const int kk    = (idx >> 5) & 127;
    const int ntile = idx >> 12;                      // 0..31
    const int half  = lane >> 4;
    const int n     = ntile * 16 + (lane & 15);
    const int k0    = kk * 4 + half * 2;
    v2f v;
    v.x = H[(size_t)k0 * HID_ + n];
    v.y = H[(size_t)(k0 + 1) * HID_ + n];
    *(v2f*)(Hf + (size_t)idx * 2) = v;
}

// ---------------------------------------------------------------------------
// Kernel 2: XI[m,n] = embedding[token(m)] @ I + b1   (m = t*B + b, 4096 x 512)
// f32 WMMA 16x16x4. One wave per 16x16 output tile; K loop over E=256.
// ---------------------------------------------------------------------------
__global__ __launch_bounds__(256) void k_embed_xi(const int*   __restrict__ x,
                                                  const float* __restrict__ emb,
                                                  const float* __restrict__ I,
                                                  const float* __restrict__ b1,
                                                  float*       __restrict__ XI) {
    const int tid   = threadIdx.x;
    const int lane  = tid & 31;
    const int wave  = tid >> 5;
    const int gid   = blockIdx.x * 8 + wave;     // 0..8191 tiles
    const int mtile = gid >> 5;                  // 0..255
    const int ntile = gid & 31;                  // 0..31
    const int half  = lane >> 4;

    const int m     = mtile * 16 + (lane & 15);  // row in [T*B]
    const int tt    = m >> 5;                    // m / B
    const int bb    = m & 31;                    // m % B
    const int token = x[bb * T_ + tt];           // input_x is [B,T]
    const float* arow = emb + (size_t)token * E_ + half * 2;
    const int n     = ntile * 16 + (lane & 15);

    v8f c = {};
    #pragma unroll 4
    for (int kk = 0; kk < E_ / 4; ++kk) {
        v2f a = *(const v2f*)(arow + kk * 4);
        int kb = kk * 4 + half * 2;
        v2f b;
        b.x = I[(size_t)kb * HID_ + n];
        b.y = I[(size_t)(kb + 1) * HID_ + n];
        c = __builtin_amdgcn_wmma_f32_16x16x4_f32(false, a, false, b,
                                                  (short)0, c, false, false);
    }
    const float bias = b1[n];
    #pragma unroll
    for (int r = 0; r < 8; ++r) {
        int mm = mtile * 16 + half * 8 + r;
        XI[(size_t)mm * HID_ + n] = c[r] + bias;
    }
}

// ---------------------------------------------------------------------------
// Kernel 3: one recurrence step  h_new = relu(h_prev @ H + XI[t])
// M=32, N=512, K=512 in f32 WMMA, B fragments from pre-swizzled Hf (b64 loads,
// 256B/wave fully coalesced).  64 tiles = 8 blocks x 8 waves.
// ---------------------------------------------------------------------------
__global__ __launch_bounds__(256) void k_rnn_step(int t,
                                                  const float* __restrict__ hprev,
                                                  float*       __restrict__ hnew,
                                                  const float* __restrict__ Hf,
                                                  const float* __restrict__ XI,
                                                  _Float16*    __restrict__ hs_h,
                                                  float*       __restrict__ last_out,
                                                  int write_last) {
    const int tid   = threadIdx.x;
    const int lane  = tid & 31;
    const int wave  = tid >> 5;
    const int gid   = blockIdx.x * 8 + wave;     // 0..63
    const int mtile = gid & 1;
    const int ntile = gid >> 1;
    const int half  = lane >> 4;

    const float* arow = hprev + (size_t)(mtile * 16 + (lane & 15)) * HID_ + half * 2;
    const float* brow = Hf + (size_t)ntile * 128 * 64 + lane * 2;
    const int n = ntile * 16 + (lane & 15);

    v8f c = {};
    #pragma unroll 4
    for (int kk = 0; kk < HID_ / 4; ++kk) {
        v2f a = *(const v2f*)(arow + kk * 4);
        v2f b = *(const v2f*)(brow + kk * 64);
        c = __builtin_amdgcn_wmma_f32_16x16x4_f32(false, a, false, b,
                                                  (short)0, c, false, false);
    }
    #pragma unroll
    for (int r = 0; r < 8; ++r) {
        int mm  = mtile * 16 + half * 8 + r;
        float v = c[r] + XI[((size_t)t * B_ + mm) * HID_ + n];
        v = fmaxf(v, 0.0f);
        hnew[(size_t)mm * HID_ + n] = v;
        hs_h[((size_t)t * B_ + mm) * HID_ + n] = (_Float16)v;
        if (write_last) last_out[(size_t)mm * HID_ + n] = v;
    }
}

// ---------------------------------------------------------------------------
// Kernel 4: logits = hs[4096,512] @ U[512,32000] + b2, scattered to [B,T,V].
// f16 WMMA 16x16x32, f32 accumulate.  WG tile = 128(M) x 64(N), 8 waves.
// B tiles DMA'd into LDS with GLOBAL_LOAD_ASYNC_TO_LDS_B128, triple-buffered
// (issue k+1 while computing k, s_wait_asynccnt, 1 barrier per k-step).
// A fragments are two contiguous b128 loads/lane from row-major f16 hs.
// ---------------------------------------------------------------------------
__global__ __launch_bounds__(256) void k_logits(const _Float16* __restrict__ hs_h,
                                                const _Float16* __restrict__ Uh,
                                                const float*    __restrict__ b2,
                                                float*          __restrict__ out) {
    __shared__ alignas(32) _Float16 ldsB[3 * 2048]; // 3 x (4 N-subtiles x 512), 12 KB
    const int tid  = threadIdx.x;
    const int lane = tid & 31;
    const int wave = tid >> 5;
    const int mblk = blockIdx.x / 500;
    const int nblk = blockIdx.x % 500;
    const int half = lane >> 4;

    const int m0 = mblk * 128 + wave * 16;
    const _Float16* abase = hs_h + (size_t)(m0 + (lane & 15)) * HID_ + half * 8;

    // cooperative B-stage mapping: 4 chunks x 64 threads x 8 halfs (16B each)
    const int cj   = tid >> 6;                   // chunk (N-subtile) 0..3
    const int coff = (tid & 63) * 8;
    const _Float16* usrc = Uh + (size_t)(nblk * 4 + cj) * 8192 + coff;
    _Float16*       ldst = ldsB + cj * 512 + coff;

    v8f acc[4] = {};

#if HAVE_ASYNC_LDS
    async_copy_b128(usrc, ldst);                 // kstep 0 -> buffer 0
    #pragma unroll
    for (int kstep = 0; kstep < 16; ++kstep) {
        const int cur = kstep % 3;
        if (kstep + 1 < 16) {
            async_copy_b128(usrc + (kstep + 1) * 512, ldst + ((kstep + 1) % 3) * 2048);
            WAIT_ASYNC(1);
        } else {
            WAIT_ASYNC(0);
        }
        __syncthreads();

        v8h alo = *(const v8h*)(abase + kstep * 32);
        v8h ahi = *(const v8h*)(abase + kstep * 32 + 16);
        v16h a;
        #pragma unroll
        for (int i = 0; i < 8; ++i) { a[i] = alo[i]; a[8 + i] = ahi[i]; }

        const v16h* bp = (const v16h*)(ldsB + cur * 2048);
        #pragma unroll
        for (int j = 0; j < 4; ++j) {
            v16h bf = bp[j * 32 + lane];
            acc[j] = __builtin_amdgcn_wmma_f32_16x16x32_f16(false, a, false, bf,
                                                            (short)0, acc[j],
                                                            false, false);
        }
    }
#else
    for (int kstep = 0; kstep < 16; ++kstep) {
        __syncthreads();
        *(v8h*)ldst = *(const v8h*)(usrc + kstep * 512);
        if (kstep + 1 < 16)
            __builtin_prefetch(usrc + (kstep + 1) * 512, 0, 1);
        __syncthreads();

        v8h alo = *(const v8h*)(abase + kstep * 32);
        v8h ahi = *(const v8h*)(abase + kstep * 32 + 16);
        v16h a;
        #pragma unroll
        for (int i = 0; i < 8; ++i) { a[i] = alo[i]; a[8 + i] = ahi[i]; }

        const v16h* bp = (const v16h*)ldsB;
        #pragma unroll
        for (int j = 0; j < 4; ++j) {
            v16h bf = bp[j * 32 + lane];
            acc[j] = __builtin_amdgcn_wmma_f32_16x16x32_f16(false, a, false, bf,
                                                            (short)0, acc[j],
                                                            false, false);
        }
    }
#endif

    #pragma unroll
    for (int r = 0; r < 8; ++r) {
        int mm = m0 + half * 8 + r;
        int tt = mm >> 5;                        // m / B
        int bb = mm & 31;                        // m % B
        float* orow = out + (size_t)bb * T_ * V_ + (size_t)tt * V_
                          + nblk * 64 + (lane & 15);
        #pragma unroll
        for (int j = 0; j < 4; ++j) {
            orow[j * 16] = acc[j][r] + b2[nblk * 64 + j * 16 + (lane & 15)];
        }
    }
}

// ---------------------------------------------------------------------------
extern "C" void kernel_launch(void* const* d_in, const int* in_sizes, int n_in,
                              void* d_out, int out_size, void* d_ws, size_t ws_size,
                              hipStream_t stream) {
    (void)in_sizes; (void)n_in; (void)out_size; (void)ws_size;
    const int*   x    = (const int*)  d_in[0];
    const float* h0in = (const float*)d_in[1];
    const float* emb  = (const float*)d_in[2];
    const float* H    = (const float*)d_in[3];
    const float* I    = (const float*)d_in[4];
    const float* b1   = (const float*)d_in[5];
    const float* U    = (const float*)d_in[6];
    const float* b2   = (const float*)d_in[7];
    float* out = (float*)d_out;

    char* ws = (char*)d_ws;
    _Float16* Uh   = (_Float16*)(ws);                                        // 32,768,000 B
    _Float16* hs_h = (_Float16*)(ws + 32768000);                             //  4,194,304 B
    float*    XI   = (float*)   (ws + 32768000 + 4194304);                   //  8,388,608 B
    float*    hA   = (float*)   (ws + 45350912);                             //     65,536 B
    float*    hB   = (float*)   (ws + 45350912 + 65536);                     //     65,536 B
    float*    Hf   = (float*)   (ws + 45350912 + 131072);                    //  1,048,576 B

    k_convert_U<<<dim3(2000), dim3(256), 0, stream>>>(U, Uh);
    k_convert_H<<<dim3(512),  dim3(256), 0, stream>>>(H, Hf);
    k_embed_xi <<<dim3(1024), dim3(256), 0, stream>>>(x, emb, I, b1, XI);

    const float* hprev = h0in;
    float* bufs[2] = {hA, hB};
    const size_t LOGITS_N = (size_t)B_ * T_ * V_;
    for (int t = 0; t < T_; ++t) {
        float* hnew = bufs[t & 1];
        k_rnn_step<<<dim3(8), dim3(256), 0, stream>>>(t, hprev, hnew, Hf, XI, hs_h,
                                                      out + LOGITS_N, t == T_ - 1);
        hprev = hnew;
    }

    k_logits<<<dim3(16000), dim3(256), 0, stream>>>(hs_h, Uh, b2, out);
}
